// BiDAF_84129819394600
// MI455X (gfx1250) — matrix operand order repaired and checked by reading, not compiled, not run on hardware
//
#include <hip/hip_runtime.h>

// ---------------- problem constants ----------------
constexpr int NB = 64;     // batch
constexpr int NT = 800;    // context length
constexpr int NJ = 50;     // query length
constexpr int ND = 200;    // feature dim
constexpr int DPAD = 224;  // ND padded to 7*32 (WMMA K-chunks)
constexpr int JPAD = 64;   // NJ padded to 4*16
constexpr int TT = 16;     // t rows per tile (WMMA M/N)
constexpr int NDT = 13;    // N-tiles of 16 covering ND (208)

typedef _Float16 f16;
typedef __attribute__((ext_vector_type(16))) _Float16 v16h;
typedef __attribute__((ext_vector_type(8)))  _Float16 v8h;
typedef __attribute__((ext_vector_type(4)))  _Float16 v4h;
typedef __attribute__((ext_vector_type(8)))  float    v8f;
typedef __attribute__((ext_vector_type(4)))  float    v4f;

__device__ __forceinline__ v16h join8(v8h lo, v8h hi) {
  return __builtin_shufflevector(lo, hi, 0,1,2,3,4,5,6,7,8,9,10,11,12,13,14,15);
}

// ---------------------------------------------------------------------------
// Tile kernel. PASS==1: row-max m[b,t] of s.  PASS==2: softmax, c2q (WMMA),
// write g = [h, c2q, h*c2q, h*q2c].  Block: 128 thr (4 waves). Grid: (50, NB).
// ---------------------------------------------------------------------------
template<int PASS>
__global__ __launch_bounds__(128)
void bidaf_tile(const float* __restrict__ h, const float* __restrict__ u,
                const float* __restrict__ w_h, const float* __restrict__ w_u,
                const float* __restrict__ w_hu,
                float* __restrict__ m_out,      // PASS 1 output
                const float* __restrict__ q2c,  // PASS 2 input
                float* __restrict__ g_out)      // PASS 2 output
{
  __shared__ alignas(32) f16   lds_hp[TT][DPAD];       // h*w_hu tile (f16)
  __shared__ alignas(32) f16   lds_up[JPAD * DPAD];    // u[j][d], later u^T[d][j]
  __shared__ float             lds_s[TT][JPAD + 1];    // s (+1 pad: no bank conflict)
  __shared__ alignas(32) f16   lds_a[TT][JPAD];        // softmax(a) f16
  __shared__ alignas(16) float lds_c2q[TT][NDT * 16];  // c2q tile
  __shared__ float             lds_su[JPAD];
  __shared__ float             lds_sh[TT];
  __shared__ alignas(16) float lds_q2cl[ND];

  const int tid  = threadIdx.x;
  const int wave = tid >> 5;
  const int lane = tid & 31;
  const int half = (lane >> 4) & 1;
  const int lm   = lane & 15;
  const int b    = blockIdx.y;
  const int t0   = blockIdx.x * TT;

  if (PASS == 1 && tid < TT) lds_sh[tid] = 0.f;
  __syncthreads();

  // ---- stage h' = h*w_hu as f16 (vectorized float4 -> 4xf16) ----
  for (int c = tid; c < TT * (DPAD / 4); c += 128) {
    int t = c / (DPAD / 4), dc = (c % (DPAD / 4)) * 4;
    v4h o = {};
    if (dc < ND) {
      v4f hv = *(const v4f*)&h[((size_t)(b*NT + t0 + t))*ND + dc];
      v4f wv = *(const v4f*)&w_hu[dc];
      #pragma unroll
      for (int i = 0; i < 4; ++i) o[i] = (f16)(hv[i] * wv[i]);
    }
    *(v4h*)&lds_hp[t][dc] = o;
  }
  // ---- stage u (j-major, f16, zero padded) ----
  for (int c = tid; c < JPAD * (DPAD / 4); c += 128) {
    int j = c / (DPAD / 4), dc = (c % (DPAD / 4)) * 4;
    v4h o = {};
    if (j < NJ && dc < ND) {
      v4f uv = *(const v4f*)&u[((size_t)(b*NJ + j))*ND + dc];
      #pragma unroll
      for (int i = 0; i < 4; ++i) o[i] = (f16)uv[i];
    }
    *(v4h*)&lds_up[j * DPAD + dc] = o;
  }
  // ---- su[j] = u[b,j,:]·w_u ----
  if (tid < JPAD) {
    float s = 0.f;
    if (tid < NJ) {
      const float* ur = &u[((size_t)(b*NJ + tid))*ND];
      for (int dc = 0; dc < ND; dc += 4) {
        v4f uv = *(const v4f*)&ur[dc];
        v4f wv = *(const v4f*)&w_u[dc];
        s += uv[0]*wv[0] + uv[1]*wv[1] + uv[2]*wv[2] + uv[3]*wv[3];
      }
    }
    lds_su[tid] = s;
  }
  if (PASS == 1) {  // sh[t] = h[b,t,:]·w_h (only the t-softmax needs it)
    int row = tid >> 3, sub = tid & 7;
    const float* hr = &h[((size_t)(b*NT + t0 + row))*ND];
    float p = 0.f;
    for (int d = sub * 4; d < ND; d += 32) {
      v4f hv = *(const v4f*)&hr[d];
      v4f wv = *(const v4f*)&w_h[d];
      p += hv[0]*wv[0] + hv[1]*wv[1] + hv[2]*wv[2] + hv[3]*wv[3];
    }
    atomicAdd(&lds_sh[row], p);
  }
  if (PASS == 2)
    for (int d = tid; d < ND; d += 128) lds_q2cl[d] = q2c[b*ND + d];
  __syncthreads();

  // ---- matmul 1 (transposed): s^T = u @ h'^T.  wave owns M-tile jt. ----
  // A = u (m=j, k=d, contiguous d);  B = h'^T (n=t, k=d, contiguous d).
  {
    const int jt = wave;  // 0..3
    v16h afr[7];
    #pragma unroll
    for (int kc = 0; kc < 7; ++kc) {
      const f16* row = &lds_up[(jt*16 + lm) * DPAD];
      v8h lo = *(const v8h*)&row[kc*32 + half*16];
      v8h hi = *(const v8h*)&row[kc*32 + half*16 + 8];
      afr[kc] = join8(lo, hi);
    }
    v8f acc = {};
    #pragma unroll
    for (int kc = 0; kc < 7; ++kc) {
      v16h bfr = *(const v16h*)&lds_hp[lm][kc*32 + half*16];
      acc = __builtin_amdgcn_wmma_f32_16x16x32_f16(false, afr[kc], false, bfr,
                                                   (short)0, acc, false, false);
    }
    #pragma unroll
    for (int r = 0; r < 8; ++r) {
      int j = jt*16 + half*8 + r;          // D elem (m=j, n=t=lm)
      lds_s[lm][j] = acc[r] + lds_su[j];
    }
  }
  __syncthreads();

  if (PASS == 1) {
    if (tid < TT) {
      float mx = -1e30f;
      for (int j = 0; j < NJ; ++j) mx = fmaxf(mx, lds_s[tid][j]);
      m_out[b*NT + t0 + tid] = mx + lds_sh[tid];
    }
    return;
  }

  // ---- refill lds_up as u^T[d][j] (global u is L2-hot) ----
  for (int c = tid; c < (JPAD / 4) * DPAD; c += 128) {
    int jc = c / DPAD, d = c % DPAD;       // jc = group of 4 j
    v4h o = {};
    if (d < ND) {
      #pragma unroll
      for (int i = 0; i < 4; ++i) {
        int j = jc*4 + i;
        if (j < NJ) o[i] = (f16)u[((size_t)(b*NJ + j))*ND + d];
      }
    }
    *(v4h*)&lds_up[d * JPAD + jc*4] = o;
  }
  // ---- softmax over j (sh & biases cancel per row) ----
  if (tid < TT) {
    float mx = -1e30f;
    for (int j = 0; j < NJ; ++j) mx = fmaxf(mx, lds_s[tid][j]);
    float sum = 0.f;
    for (int j = 0; j < NJ; ++j) sum += __expf(lds_s[tid][j] - mx);
    float inv = 1.f / sum;
    for (int j = 0; j < JPAD; ++j) {
      float a = (j < NJ) ? __expf(lds_s[tid][j] - mx) * inv : 0.f;
      lds_a[tid][j] = (f16)a;
    }
  }
  __syncthreads();

  // ---- matmul 2: c2q = a @ u (A = a[t][j], B = u^T[d][j], both contiguous) --
  v16h aa[2];
  #pragma unroll
  for (int kc = 0; kc < 2; ++kc) {
    v8h lo = *(const v8h*)&lds_a[lm][kc*32 + half*16];
    v8h hi = *(const v8h*)&lds_a[lm][kc*32 + half*16 + 8];
    aa[kc] = join8(lo, hi);
  }
  for (int nt = wave; nt < NDT; nt += 4) {
    v8f acc = {};
    #pragma unroll
    for (int kc = 0; kc < 2; ++kc) {
      v16h bfr = *(const v16h*)&lds_up[(nt*16 + lm) * JPAD + kc*32 + half*16];
      acc = __builtin_amdgcn_wmma_f32_16x16x32_f16(false, aa[kc], false, bfr,
                                                   (short)0, acc, false, false);
    }
    #pragma unroll
    for (int r = 0; r < 8; ++r)
      lds_c2q[half*8 + r][nt*16 + lm] = acc[r];
  }
  __syncthreads();

  // ---- write g = [h, c2q, h*c2q, h*q2c]: float4, non-temporal, coalesced ----
  for (int c = tid; c < TT * (ND / 4); c += 128) {
    int t = c / (ND / 4), dc = (c % (ND / 4)) * 4;
    v4f hv = *(const v4f*)&h[((size_t)(b*NT + t0 + t))*ND + dc];
    v4f cv = *(const v4f*)&lds_c2q[t][dc];
    v4f qv = *(const v4f*)&lds_q2cl[dc];
    v4f hc, hq;
    #pragma unroll
    for (int i = 0; i < 4; ++i) { hc[i] = hv[i]*cv[i]; hq[i] = hv[i]*qv[i]; }
    size_t base = ((size_t)(b*NT + t0 + t)) * (4*ND);
    __builtin_nontemporal_store(hv, (v4f*)&g_out[base + dc]);
    __builtin_nontemporal_store(cv, (v4f*)&g_out[base + ND + dc]);
    __builtin_nontemporal_store(hc, (v4f*)&g_out[base + 2*ND + dc]);
    __builtin_nontemporal_store(hq, (v4f*)&g_out[base + 3*ND + dc]);
  }
}

// ---------------------------------------------------------------------------
// q2c kernel: per batch, softmax(m) over T then q2c[b,d] = sum_t w_t h[b,t,d].
// ---------------------------------------------------------------------------
__global__ __launch_bounds__(256)
void bidaf_q2c(const float* __restrict__ m, const float* __restrict__ h,
               float* __restrict__ q2c)
{
  __shared__ float red[256];
  __shared__ float wts[NT];
  const int tid = threadIdx.x;
  const int b   = blockIdx.x;

  float pmax = -1e30f;
  for (int t = tid; t < NT; t += 256) pmax = fmaxf(pmax, m[b*NT + t]);
  red[tid] = pmax;
  __syncthreads();
  for (int s = 128; s > 0; s >>= 1) {
    if (tid < s) red[tid] = fmaxf(red[tid], red[tid + s]);
    __syncthreads();
  }
  float mx = red[0];
  __syncthreads();

  float psum = 0.f;
  for (int t = tid; t < NT; t += 256) {
    float e = __expf(m[b*NT + t] - mx);
    wts[t] = e;
    psum += e;
  }
  red[tid] = psum;
  __syncthreads();
  for (int s = 128; s > 0; s >>= 1) {
    if (tid < s) red[tid] += red[tid + s];
    __syncthreads();
  }
  float inv = 1.f / red[0];

  for (int d = tid; d < ND; d += 256) {
    float acc = 0.f;
    for (int t = 0; t < NT; ++t)
      acc += wts[t] * h[((size_t)(b*NT + t))*ND + d];   // coalesced across tid
    q2c[b*ND + d] = acc * inv;
  }
}

// ---------------------------------------------------------------------------
extern "C" void kernel_launch(void* const* d_in, const int* in_sizes, int n_in,
                              void* d_out, int out_size, void* d_ws, size_t ws_size,
                              hipStream_t stream) {
  (void)in_sizes; (void)n_in; (void)out_size; (void)ws_size;
  const float* h    = (const float*)d_in[0];
  const float* u    = (const float*)d_in[1];
  const float* w_h  = (const float*)d_in[2];
  // d_in[3]=b_h, d_in[5]=b_u, d_in[7]=b_hu: additive constants cancel in both
  // softmaxes and never affect g -> not read.
  const float* w_u  = (const float*)d_in[4];
  const float* w_hu = (const float*)d_in[6];

  float* m_ws   = (float*)d_ws;          // [NB*NT]
  float* q2c_ws = m_ws + NB*NT;          // [NB*ND]
  float* g      = (float*)d_out;         // [NB, NT, 4*ND]

  dim3 grid(NT / TT, NB);
  bidaf_tile<1><<<grid, 128, 0, stream>>>(h, u, w_h, w_u, w_hu,
                                          m_ws, nullptr, nullptr);
  bidaf_q2c<<<NB, 256, 0, stream>>>(m_ws, h, q2c_ws);
  bidaf_tile<2><<<grid, 128, 0, stream>>>(h, u, w_h, w_u, w_hu,
                                          nullptr, q2c_ws, g);
}
